// CrossSparseAggrNet_v2_730144441135
// MI455X (gfx1250) — matrix operand order repaired and checked by reading, not compiled, not run on hardware
//
#include <hip/hip_runtime.h>
#include <math.h>

// ---------------- CDNA5 types ----------------
typedef __attribute__((ext_vector_type(16))) __bf16 v16bf;
typedef __attribute__((ext_vector_type(8)))  float  v8f;

// ---------------- problem dims ----------------
#define BV    128
#define LV    392
#define CDIM  512
#define LT    50
#define HDIM  102      // int(512*0.2)
#define KDIM  117      // int(392*0.5*0.6)
#define NKEEP 236      // ceil(392*0.6)
#define NNON  156      // 392-236
#define OUTR  118      // 117 aggr rows + 1 extra

// padded GEMM dims (multiples of 16 / 32)
#define MPAD  240      // kept rows padded      (15 M-tiles)
#define NPAD1 128      // H padded              (8 N-tiles; cols>=102 zero)
#define KPAD2 128      // H padded for GEMM2 K  (4 K-tiles)
#define NPAD2 128      // KDIM padded           (8 N-tiles)
#define KPAD3 256      // kept rows padded as K (8 K-tiles)

#define TM1 15
#define TN1 8
#define TK1 16   // 512/32
#define TM2 15
#define TN2 8
#define TK2 4    // 128/32
#define TM3 8    // 128/16
#define TN3 32   // 512/16
#define TK3 8    // 256/32

#define NB  4    // N-tiles per wave (4 accumulators: 16x64 strip)
#define NG1 (TN1/NB)
#define NG2 (TN2/NB)
#define NG3 (TN3/NB)

// per-batch strides, in bf16 halves / floats
#define A1_STRIDE  (TM1*TK1*512)          // 122880 halves
#define SB_STRIDE  (TK3*TN3*512)          // 131072 halves
#define A2_STRIDE  (TM2*TK2*512)          // 30720  halves
#define A3_STRIDE  (TM3*TK3*512)          // 32768  halves
#define TT_STRIDE  (NPAD2*MPAD)           // 30720  floats

// workspace byte offsets (all 32B-aligned)
#define OFF_GLO    0u
#define OFF_SCORE  2048u
#define OFF_ORDER  202752u
#define OFF_NW     403456u
#define OFF_W1P    483328u          // 16*8*512*2   = 131072 B
#define OFF_W2P    614400u          // 4*8*512*2    = 32768 B
#define OFF_A1     647168u          // 128*122880*2 = 31457280 B
#define OFF_SELB   32104448u        // 128*131072*2 = 33554432 B
#define OFF_A2     65658880u        // 128*30720*2  = 7864320 B
#define OFF_TT     73523200u        // 128*30720*4  = 15728640 B
#define OFF_A3     89251840u        // 128*32768*2  = 8388608 B

// ---------------- helpers ----------------
__device__ __forceinline__ unsigned short f2bf(float f) {
  union { float f; unsigned u; } v; v.f = f;
  unsigned r = v.u + 0x7FFFu + ((v.u >> 16) & 1u);   // round-to-nearest-even
  return (unsigned short)(r >> 16);
}

__device__ __forceinline__ float wave_red_add(float x) {
  #pragma unroll
  for (int m = 16; m > 0; m >>= 1) x += __shfl_xor(x, m, 32);
  return x;
}

// A-operand (MxK) bf16 fragment packing: 16x32 tiles, contiguous k order (k = 16*(lane/16) + element)
__device__ __forceinline__ int a_pack_idx(int nKt, int m, int k) {
  int tm = m >> 4, tk = k >> 5;
  int koff = k & 31;
  int lane = (m & 15) | ((koff >> 4) << 4);   // hi-lanes hold K 16..31
  int h    = koff & 15;
  return (((tm * nKt + tk) << 5) + lane) * 16 + h;
}
// B-operand (KxN) bf16 fragment packing: 32x16 tiles, contiguous k order (k = 16*(lane/16) + element)
__device__ __forceinline__ int b_pack_idx(int nNt, int k, int n) {
  int tk = k >> 5, tn = n >> 4;
  int koff = k & 31;
  int lane = (n & 15) | ((koff >> 4) << 4);   // hi-lanes hold K 16..31
  int h    = koff & 15;
  return (((tk * nNt + tn) << 5) + lane) * 16 + h;
}

#define WMMA_BF16(A, B, C) \
  __builtin_amdgcn_wmma_f32_16x16x32_bf16(false, (A), false, (B), (short)0, (C), false, false)

// Async-stage a B K-strip (nKt x NB tiles of 1KB) into LDS; one wave's worth of
// ASYNCcnt loads per thread, then drain ASYNCcnt. Caller must __syncthreads().
__device__ __forceinline__ void async_stage_b(const unsigned short* __restrict__ gbase,
                                              unsigned lds_base, int nKt, int nNt,
                                              int ng, int tid) {
  int nchunk = nKt * NB * 64;                 // 16B chunks
  for (int c = tid; c < nchunk; c += 256) {
    int tile = c >> 6;
    int kt = tile >> 2, j = tile & 3;
    int gtile = kt * nNt + ng * NB + j;
    const unsigned short* src = gbase + ((size_t)gtile << 9) + ((size_t)(c & 63) << 3);
    unsigned long long ga = (unsigned long long)(const void*)src;
    unsigned la = lds_base + ((unsigned)c << 4);
    asm volatile("global_load_async_to_lds_b128 %0, %1, off"
                 :: "v"(la), "v"(ga) : "memory");
  }
  asm volatile("s_wait_asynccnt 0" ::: "memory");
}

// ---------------- K0a: cap_glo = l2norm(mean of cap_embs[-1])  ----------------
__global__ void k_glo(const float* __restrict__ cap, float* __restrict__ glo) {
  __shared__ float sred[512];
  int c = threadIdx.x;
  const float* capb = cap + (size_t)(BV - 1) * LT * CDIM;
  float s = 0.f;
  for (int l = 0; l < LT; ++l) s += capb[l * CDIM + c];
  s /= (float)LT;
  sred[c] = s * s;
  __syncthreads();
  for (int st = 256; st > 0; st >>= 1) {
    if (c < st) sred[c] += sred[c + st];
    __syncthreads();
  }
  float nm = fmaxf(sqrtf(sred[0]), 1e-12f);
  glo[c] = s / nm;
}

// ---------------- K0b/K0c: pack weights into B-fragment layout ----------------
__global__ void k_pack_w1(const float* __restrict__ w1, unsigned short* __restrict__ w1p) {
  int id = blockIdx.x * 256 + threadIdx.x;
  if (id >= TK1 * TN1 * 512) return;
  int h = id & 15, lane = (id >> 4) & 31, tile = id >> 9;
  int tn = tile % TN1, tk = tile / TN1;
  int k = h + ((lane >> 4) << 4) + (tk << 5);
  int n = (lane & 15) + (tn << 4);
  w1p[id] = (n < HDIM) ? f2bf(w1[k * HDIM + n]) : (unsigned short)0;
}
__global__ void k_pack_w2(const float* __restrict__ w2, unsigned short* __restrict__ w2p) {
  int id = blockIdx.x * 256 + threadIdx.x;
  if (id >= TK2 * TN2 * 512) return;
  int h = id & 15, lane = (id >> 4) & 31, tile = id >> 9;
  int tn = tile % TN2, tk = tile / TN2;
  int k = h + ((lane >> 4) << 4) + (tk << 5);
  int n = (lane & 15) + (tn << 4);
  w2p[id] = (k < HDIM && n < KDIM) ? f2bf(w2[k * KDIM + n]) : (unsigned short)0;
}

// ---------------- K1: token scores ----------------
__global__ void k_score(const float* __restrict__ img, const float* __restrict__ glo,
                        float* __restrict__ score) {
  __shared__ float smean[CDIM];
  __shared__ float sglo[CDIM];
  __shared__ float sred[256];
  __shared__ float s_nm;
  int b = blockIdx.x, tid = threadIdx.x;
  const float* imgb = img + (size_t)b * LV * CDIM;
  for (int c = tid; c < CDIM; c += 256) {
    float s = 0.f;
    for (int l = 0; l < LV; ++l) s += imgb[(size_t)l * CDIM + c];
    smean[c] = s / (float)LV;
    sglo[c] = glo[c];
  }
  __syncthreads();
  float q = 0.f;
  for (int c = tid; c < CDIM; c += 256) q += smean[c] * smean[c];
  sred[tid] = q;
  __syncthreads();
  for (int st = 128; st > 0; st >>= 1) {
    if (tid < st) sred[tid] += sred[tid + st];
    __syncthreads();
  }
  if (tid == 0) s_nm = fmaxf(sqrtf(sred[0]), 1e-12f);
  __syncthreads();
  int wid = tid >> 5, lane = tid & 31;
  for (int l = wid; l < LV; l += 8) {
    const float* row = imgb + (size_t)l * CDIM;
    float dm = 0.f, dg = 0.f, dx = 0.f;
    for (int c = lane; c < CDIM; c += 32) {
      float x = row[c];
      dm += x * smean[c];
      dg += x * sglo[c];
      dx += x * x;
    }
    dm = wave_red_add(dm);
    dg = wave_red_add(dg);
    dx = wave_red_add(dx);
    if (lane == 0) {
      float nx = fmaxf(sqrtf(dx), 1e-12f);
      score[b * LV + l] = (dm / s_nm + dg) / nx;
    }
  }
}

// ---------------- K2: bitonic argsort (desc) + tail softmax weights ----------------
__global__ void k_sort(const float* __restrict__ score, int* __restrict__ order,
                       float* __restrict__ nw) {
  __shared__ float sk[512];
  __shared__ int   si[512];
  __shared__ float sred[256];
  int b = blockIdx.x, tid = threadIdx.x;
  for (int t = tid; t < 512; t += 256) {
    sk[t] = (t < LV) ? score[b * LV + t] : -3.402823466e38f;
    si[t] = t;
  }
  __syncthreads();
  for (int k = 2; k <= 512; k <<= 1) {
    for (int j = k >> 1; j > 0; j >>= 1) {
      for (int t = tid; t < 512; t += 256) {
        int x = t ^ j;
        if (x > t) {
          bool dir = ((t & k) == 0);   // descending overall
          float a = sk[t], c = sk[x];
          if ((a < c) == dir) {
            sk[t] = c; sk[x] = a;
            int ti = si[t]; si[t] = si[x]; si[x] = ti;
          }
        }
      }
      __syncthreads();
    }
  }
  for (int t = tid; t < LV; t += 256) order[b * LV + t] = si[t];
  // softmax over dropped scores sk[236..391] (sorted desc -> sk[236] is the max)
  float mx = sk[NKEEP];
  float e = (tid < NNON) ? expf(sk[NKEEP + tid] - mx) : 0.f;
  sred[tid] = e;
  __syncthreads();
  for (int st = 128; st > 0; st >>= 1) {
    if (tid < st) sred[tid] += sred[tid + st];
    __syncthreads();
  }
  float inv = 1.f / sred[0];
  if (tid < NNON) nw[b * NNON + tid] = e * inv;
}

// ---------------- K3: extra = softmax-weighted sum of dropped rows ----------------
__global__ void k_extra(const float* __restrict__ img, const int* __restrict__ order,
                        const float* __restrict__ nw, float* __restrict__ out) {
  __shared__ float wsh[NNON];
  __shared__ int   osh[NNON];
  int b = blockIdx.x, tid = threadIdx.x;
  for (int j = tid; j < NNON; j += 256) {
    wsh[j] = nw[b * NNON + j];
    osh[j] = order[b * LV + NKEEP + j];
  }
  __syncthreads();
  const float* imgb = img + (size_t)b * LV * CDIM;
  for (int c = tid; c < CDIM; c += 256) {
    float s = 0.f;
    for (int j = 0; j < NNON; ++j) s += wsh[j] * imgb[(size_t)osh[j] * CDIM + c];
    out[((size_t)b * OUTR + (OUTR - 1)) * CDIM + c] = s;
  }
}

// ---------------- K4: gather kept rows, LayerNorm -> A1 frags; raw sel -> B frags ----------------
__global__ void k_ln_stage(const float* __restrict__ img, const int* __restrict__ order,
                           const float* __restrict__ g, const float* __restrict__ bb,
                           unsigned short* __restrict__ A1, unsigned short* __restrict__ SB) {
  __shared__ float sr1[256], sr2[256];
  __shared__ float s_mu, s_rs;
  int b = blockIdx.x, tid = threadIdx.x;
  const int* ordb = order + b * LV;
  unsigned short* A1b = A1 + (size_t)b * A1_STRIDE;
  unsigned short* SBb = SB + (size_t)b * SB_STRIDE;
  int c0 = tid, c1 = tid + 256;
  for (int i = 0; i < KPAD3; ++i) {
    if (i < NKEEP) {
      const float* src = img + ((size_t)b * LV + ordb[i]) * CDIM;
      float x0 = src[c0], x1 = src[c1];
      sr1[tid] = x0 + x1;
      sr2[tid] = x0 * x0 + x1 * x1;
      __syncthreads();
      for (int st = 128; st > 0; st >>= 1) {
        if (tid < st) { sr1[tid] += sr1[tid + st]; sr2[tid] += sr2[tid + st]; }
        __syncthreads();
      }
      if (tid == 0) {
        float mu = sr1[0] * (1.f / 512.f);
        float var = sr2[0] * (1.f / 512.f) - mu * mu;
        s_mu = mu; s_rs = rsqrtf(var + 1e-5f);
      }
      __syncthreads();
      float mu = s_mu, rs = s_rs;
      A1b[a_pack_idx(TK1, i, c0)] = f2bf((x0 - mu) * rs * g[c0] + bb[c0]);
      A1b[a_pack_idx(TK1, i, c1)] = f2bf((x1 - mu) * rs * g[c1] + bb[c1]);
      SBb[b_pack_idx(TN3, i, c0)] = f2bf(x0);
      SBb[b_pack_idx(TN3, i, c1)] = f2bf(x1);
    } else {
      if (i < MPAD) {
        A1b[a_pack_idx(TK1, i, c0)] = 0;
        A1b[a_pack_idx(TK1, i, c1)] = 0;
      }
      SBb[b_pack_idx(TN3, i, c0)] = 0;
      SBb[b_pack_idx(TN3, i, c1)] = 0;
    }
  }
}

// ---------------- K5: GEMM1, B strip async-staged in LDS ----------------
__global__ void k_gemm1(const unsigned short* __restrict__ A1,
                        const unsigned short* __restrict__ w1p,
                        const float* __restrict__ b1,
                        unsigned short* __restrict__ A2) {
  __shared__ __align__(32) unsigned short sB[TK1 * NB * 512];   // 64 KB
  int b = blockIdx.x / NG1, ng = blockIdx.x % NG1;
  int tid = threadIdx.x, wave = tid >> 5, lane = tid & 31;
  async_stage_b(w1p, (unsigned)(unsigned long long)(const void*)sB, TK1, TN1, ng, tid);
  __syncthreads();
  const unsigned short* Abase = A1 + (size_t)b * A1_STRIDE + (lane << 4);
  unsigned short* A2b = A2 + (size_t)b * A2_STRIDE;
  for (int rep = 0; rep < 2; ++rep) {
    int tm = rep * 8 + wave;
    if (tm >= TM1) break;
    const unsigned short* Ab = Abase + ((size_t)tm * TK1 << 9);
    v8f acc[NB] = {};
    for (int kt = 0; kt < TK1; ++kt) {
      v16bf a = *(const v16bf*)(Ab + ((size_t)kt << 9));
      if (kt + 1 < TK1) __builtin_prefetch(Ab + ((size_t)(kt + 1) << 9), 0, 0);
      const unsigned short* bk = sB + ((kt * NB) << 9) + (lane << 4);
      v16bf w0 = *(const v16bf*)(bk);
      v16bf w1x = *(const v16bf*)(bk + 512);
      v16bf w2x = *(const v16bf*)(bk + 1024);
      v16bf w3x = *(const v16bf*)(bk + 1536);
      acc[0] = WMMA_BF16(a, w0, acc[0]);
      acc[1] = WMMA_BF16(a, w1x, acc[1]);
      acc[2] = WMMA_BF16(a, w2x, acc[2]);
      acc[3] = WMMA_BF16(a, w3x, acc[3]);
    }
    int mbase = (tm << 4) + ((lane >> 4) << 3);
    #pragma unroll
    for (int j = 0; j < NB; ++j) {
      int n = ((ng * NB + j) << 4) + (lane & 15);
      float bias = (n < HDIM) ? b1[n] : 0.f;
      #pragma unroll
      for (int v = 0; v < 8; ++v) {
        float x = acc[j][v] + bias;
        float gl = 0.5f * x * (1.f + erff(x * 0.70710678118654752f));  // exact GELU
        A2b[a_pack_idx(TK2, mbase + v, n)] = f2bf(gl);
      }
    }
  }
}

// ---------------- K6: GEMM2, B strip async-staged in LDS ----------------
__global__ void k_gemm2(const unsigned short* __restrict__ A2,
                        const unsigned short* __restrict__ w2p,
                        const float* __restrict__ b2,
                        const float* __restrict__ scale,
                        float* __restrict__ Tt) {
  __shared__ __align__(32) unsigned short sB[TK2 * NB * 512];   // 16 KB
  int b = blockIdx.x / NG2, ng = blockIdx.x % NG2;
  int tid = threadIdx.x, wave = tid >> 5, lane = tid & 31;
  async_stage_b(w2p, (unsigned)(unsigned long long)(const void*)sB, TK2, TN2, ng, tid);
  __syncthreads();
  const unsigned short* Abase = A2 + (size_t)b * A2_STRIDE + (lane << 4);
  float sc = scale[0];
  for (int rep = 0; rep < 2; ++rep) {
    int tm = rep * 8 + wave;
    if (tm >= TM2) break;
    const unsigned short* Ab = Abase + ((size_t)tm * TK2 << 9);
    v8f acc[NB] = {};
    #pragma unroll
    for (int kt = 0; kt < TK2; ++kt) {
      v16bf a = *(const v16bf*)(Ab + ((size_t)kt << 9));
      const unsigned short* bk = sB + ((kt * NB) << 9) + (lane << 4);
      v16bf w0 = *(const v16bf*)(bk);
      v16bf w1x = *(const v16bf*)(bk + 512);
      v16bf w2x = *(const v16bf*)(bk + 1024);
      v16bf w3x = *(const v16bf*)(bk + 1536);
      acc[0] = WMMA_BF16(a, w0, acc[0]);
      acc[1] = WMMA_BF16(a, w1x, acc[1]);
      acc[2] = WMMA_BF16(a, w2x, acc[2]);
      acc[3] = WMMA_BF16(a, w3x, acc[3]);
    }
    int mbase = (tm << 4) + ((lane >> 4) << 3);
    #pragma unroll
    for (int j = 0; j < NB; ++j) {
      int n = ((ng * NB + j) << 4) + (lane & 15);
      float bias = (n < KDIM) ? b2[n] : 0.f;
      float* row = Tt + ((size_t)b * NPAD2 + n) * MPAD;
      #pragma unroll
      for (int v = 0; v < 8; ++v) row[mbase + v] = (acc[j][v] + bias) * sc;
    }
  }
}

// ---------------- K7: column softmax over kept rows -> A3 frags ----------------
__global__ void k_softmax(const float* __restrict__ Tt, unsigned short* __restrict__ A3) {
  int b = blockIdx.x, n = threadIdx.x;   // 128 threads, one column each
  const float* tt = Tt + ((size_t)b * NPAD2 + n) * MPAD;
  unsigned short* A3b = A3 + (size_t)b * A3_STRIDE;
  if (n < KDIM) {
    float mx = tt[0];
    for (int m = 1; m < NKEEP; ++m) mx = fmaxf(mx, tt[m]);
    float s = 0.f;
    for (int m = 0; m < NKEEP; ++m) s += expf(tt[m] - mx);
    float inv = 1.f / s;
    for (int m = 0; m < KPAD3; ++m) {
      float w = (m < NKEEP) ? expf(tt[m] - mx) * inv : 0.f;
      A3b[a_pack_idx(TK3, n, m)] = f2bf(w);
    }
  } else {
    for (int m = 0; m < KPAD3; ++m) A3b[a_pack_idx(TK3, n, m)] = 0;
  }
}

// ---------------- K8: GEMM3, per-batch sel B strip async-staged in LDS ----------------
__global__ void k_gemm3(const unsigned short* __restrict__ A3,
                        const unsigned short* __restrict__ SB,
                        float* __restrict__ out) {
  __shared__ __align__(32) unsigned short sB[TK3 * NB * 512];   // 32 KB
  int b = blockIdx.x / NG3, ng = blockIdx.x % NG3;
  int tid = threadIdx.x, wave = tid >> 5, lane = tid & 31;
  async_stage_b(SB + (size_t)b * SB_STRIDE,
                (unsigned)(unsigned long long)(const void*)sB, TK3, TN3, ng, tid);
  __syncthreads();
  int tm = wave;   // 8 waves = 8 M-tiles
  const unsigned short* Ab = A3 + (size_t)b * A3_STRIDE + ((size_t)tm * TK3 << 9) + (lane << 4);
  v8f acc[NB] = {};
  for (int kt = 0; kt < TK3; ++kt) {
    v16bf a = *(const v16bf*)(Ab + ((size_t)kt << 9));
    if (kt + 1 < TK3) __builtin_prefetch(Ab + ((size_t)(kt + 1) << 9), 0, 0);
    const unsigned short* bk = sB + ((kt * NB) << 9) + (lane << 4);
    v16bf w0 = *(const v16bf*)(bk);
    v16bf w1x = *(const v16bf*)(bk + 512);
    v16bf w2x = *(const v16bf*)(bk + 1024);
    v16bf w3x = *(const v16bf*)(bk + 1536);
    acc[0] = WMMA_BF16(a, w0, acc[0]);
    acc[1] = WMMA_BF16(a, w1x, acc[1]);
    acc[2] = WMMA_BF16(a, w2x, acc[2]);
    acc[3] = WMMA_BF16(a, w3x, acc[3]);
  }
  int mbase = (tm << 4) + ((lane >> 4) << 3);
  #pragma unroll
  for (int j = 0; j < NB; ++j) {
    int n = ((ng * NB + j) << 4) + (lane & 15);
    #pragma unroll
    for (int v = 0; v < 8; ++v) {
      int m = mbase + v;
      if (m < KDIM) out[((size_t)b * OUTR + m) * CDIM + n] = acc[j][v];
    }
  }
}

// ---------------- host launch ----------------
extern "C" void kernel_launch(void* const* d_in, const int* in_sizes, int n_in,
                              void* d_out, int out_size, void* d_ws, size_t ws_size,
                              hipStream_t stream) {
  (void)in_sizes; (void)n_in; (void)out_size; (void)ws_size;
  const float* img  = (const float*)d_in[0];
  const float* cap  = (const float*)d_in[1];
  const float* ln_g = (const float*)d_in[2];
  const float* ln_b = (const float*)d_in[3];
  const float* w1   = (const float*)d_in[4];
  const float* b1   = (const float*)d_in[5];
  const float* w2   = (const float*)d_in[6];
  const float* b2   = (const float*)d_in[7];
  const float* sc   = (const float*)d_in[8];
  float* out = (float*)d_out;

  char* ws = (char*)d_ws;
  float*          glo   = (float*)(ws + OFF_GLO);
  float*          score = (float*)(ws + OFF_SCORE);
  int*            order = (int*)(ws + OFF_ORDER);
  float*          nw    = (float*)(ws + OFF_NW);
  unsigned short* w1p   = (unsigned short*)(ws + OFF_W1P);
  unsigned short* w2p   = (unsigned short*)(ws + OFF_W2P);
  unsigned short* A1    = (unsigned short*)(ws + OFF_A1);
  unsigned short* SB    = (unsigned short*)(ws + OFF_SELB);
  unsigned short* A2    = (unsigned short*)(ws + OFF_A2);
  float*          Tt    = (float*)(ws + OFF_TT);
  unsigned short* A3    = (unsigned short*)(ws + OFF_A3);

  k_glo<<<1, 512, 0, stream>>>(cap, glo);
  k_pack_w1<<<(TK1 * TN1 * 512 + 255) / 256, 256, 0, stream>>>(w1, w1p);
  k_pack_w2<<<(TK2 * TN2 * 512 + 255) / 256, 256, 0, stream>>>(w2, w2p);
  k_score<<<BV, 256, 0, stream>>>(img, glo, score);
  k_sort<<<BV, 256, 0, stream>>>(score, order, nw);
  k_extra<<<BV, 256, 0, stream>>>(img, order, nw, out);
  k_ln_stage<<<BV, 256, 0, stream>>>(img, order, ln_g, ln_b, A1, SB);
  k_gemm1<<<BV * NG1, 256, 0, stream>>>(A1, w1p, b1, A2);
  k_gemm2<<<BV * NG2, 256, 0, stream>>>(A2, w2p, b2, sc, Tt);
  k_softmax<<<BV, 128, 0, stream>>>(Tt, A3);
  k_gemm3<<<BV * NG3, 256, 0, stream>>>(A3, SB, out);
}
